// lightgcl_frame_bsl_81432579932608
// MI455X (gfx1250) — matrix operand order repaired and checked by reading, not compiled, not run on hardware
//
// LightGCL forward for MI455X (gfx1250, wave32).
// Heavy GEMMs (G[uids] @ E^T fused exp-sum, and u_e @ p_e^T) use
// v_wmma_f32_16x16x32_f16 (K=64 => 2 WMMA per 16x16 tile, f32 accum).
// SpMM uses f32 global atomics (random scatter, HBM-bound). Low-rank SVD path
// reduced to two 5x64 temp matrices (G never materialized). ~102 MB workspace.
#include <hip/hip_runtime.h>

typedef __attribute__((ext_vector_type(16))) _Float16 v16h;
typedef __attribute__((ext_vector_type(8)))  _Float16 v8h;
typedef __attribute__((ext_vector_type(8)))  float    v8f;

static constexpr int   kUsers = 100000;
static constexpr int   kItems = 50000;
static constexpr int   kDim   = 64;
static constexpr int   kRank  = 5;
static constexpr int   kNnz   = 1000000;
static constexpr int   kBatch = 1024;
static constexpr float kTemp  = 0.2f;
static constexpr float kT1    = 0.1f;
static constexpr float kT2    = 1.0f;
static constexpr float kL1    = 0.2f;
static constexpr float kL2    = 1e-7f;

// ---------------------------------------------------------------- zero
__global__ void k_zero(float* __restrict__ p, size_t n) {
  size_t i  = (size_t)blockIdx.x * blockDim.x + threadIdx.x;
  size_t st = (size_t)gridDim.x * blockDim.x;
  for (; i < n; i += st) p[i] = 0.0f;
}

// ---------------------------------------------------------------- SpMM (both directions in one pass)
// z_u[rows[k],:] += v * e_i[cols[k],:]   ;   z_i[cols[k],:] += v * e_u[rows[k],:]
__global__ void k_spmm(const float* __restrict__ eu, const float* __restrict__ ei,
                       const int* __restrict__ rows, const int* __restrict__ cols,
                       const float* __restrict__ vals,
                       float* __restrict__ zu, float* __restrict__ zi) {
  long idx = (long)blockIdx.x * blockDim.x + threadIdx.x;
  if (idx >= (long)kNnz * kDim) return;
  int k = (int)(idx >> 6);
  int d = (int)(idx & 63);
  int r = rows[k], c = cols[k];
  float v = vals[k];
  atomicAdd(&zu[(size_t)r * kDim + d], v * ei[(size_t)c * kDim + d]);
  atomicAdd(&zi[(size_t)c * kDim + d], v * eu[(size_t)r * kDim + d]);
}

// ---------------------------------------------------------------- rank-5 reduction: tmp[5][64] = pt @ (E0 + z)
__global__ void k_rank(const float* __restrict__ pt, const float* __restrict__ E0,
                       const float* __restrict__ z, float* __restrict__ tmp, int N) {
  __shared__ float lds[kRank * kDim];
  const int tid = threadIdx.x;
  for (int i = tid; i < kRank * kDim; i += blockDim.x) lds[i] = 0.0f;
  __syncthreads();
  const int d   = tid & 63;
  const int sub = tid >> 6;               // 4 rows in flight per block
  float a[kRank] = {};
  for (long i = (long)blockIdx.x * 4 + sub; i < N; i += (long)gridDim.x * 4) {
    float x = E0[i * kDim + d] + z[i * kDim + d];
#pragma unroll
    for (int r = 0; r < kRank; ++r) a[r] += pt[(size_t)r * N + i] * x;
  }
#pragma unroll
  for (int r = 0; r < kRank; ++r) atomicAdd(&lds[r * kDim + d], a[r]);
  __syncthreads();
  for (int i = tid; i < kRank * kDim; i += blockDim.x) atomicAdd(&tmp[i], lds[i]);
}

// ---------------------------------------------------------------- E_final = E0 + zA + zB (in-place into zB) + f16 mirror
__global__ void k_combine(const float* __restrict__ E0, const float* __restrict__ zA,
                          float* __restrict__ zB, _Float16* __restrict__ e16, size_t n) {
  size_t i  = (size_t)blockIdx.x * blockDim.x + threadIdx.x;
  size_t st = (size_t)gridDim.x * blockDim.x;
  for (; i < n; i += st) {
    float v = E0[i] + zA[i] + zB[i];
    zB[i] = v;
    e16[i] = (_Float16)v;
  }
}

// ---------------------------------------------------------------- gather sampled G rows: G[id] = E0[id] + mul_s[id] @ tmp
__global__ void __launch_bounds__(64)
k_gather(const float* __restrict__ E0, const float* __restrict__ muls,
         const float* __restrict__ tmp, const int* __restrict__ idxA,
         const int* __restrict__ idxB, float* __restrict__ Gs,
         _Float16* __restrict__ G16) {
  int b = blockIdx.x, d = threadIdx.x;
  int id = (idxB != nullptr && b >= kBatch) ? idxB[b - kBatch] : idxA[b];
  float v = E0[(size_t)id * kDim + d];
#pragma unroll
  for (int r = 0; r < kRank; ++r) v += muls[(size_t)id * kRank + r] * tmp[r * kDim + d];
  Gs[(size_t)b * kDim + d]  = v;
  G16[(size_t)b * kDim + d] = (_Float16)v;
}

// ---------------------------------------------------------------- WMMA helpers (contiguous k order, same in A and B)
__device__ __forceinline__ v8f wmma_k32(v16h a, v16h b, v8f c) {
  return __builtin_amdgcn_wmma_f32_16x16x32_f16(false, a, false, b, (short)0, c, false, false);
}
// A 16x32 f16 operand: lane l<16 -> row m=l, K0..15; lanes 16..31 -> K16..31 (contiguous)
__device__ __forceinline__ void load_a64(const _Float16* __restrict__ arow, int half,
                                         v16h& a0, v16h& a1) {
  v8h lo0 = *(const v8h*)(arow + half * 16);
  v8h hi0 = *(const v8h*)(arow + 8 + half * 16);
  v8h lo1 = *(const v8h*)(arow + 32 + half * 16);
  v8h hi1 = *(const v8h*)(arow + 40 + half * 16);
#pragma unroll
  for (int i = 0; i < 8; ++i) {
    a0[i] = lo0[i]; a0[i + 8] = hi0[i];
    a1[i] = lo1[i]; a1[i + 8] = hi1[i];
  }
}

// ---------------------------------------------------------------- fused GEMM + exp-row-sum:
// S[m] += sum_n exp( dot(A16[m,:], B16[n,:]) / TEMP )   over this block's n-chunk
__global__ void __launch_bounds__(32)
k_expsum(const _Float16* __restrict__ A16, const _Float16* __restrict__ B16,
         float* __restrict__ S, int tilesPerChunk) {
  const int lane = threadIdx.x;
  const int half = lane >> 4;
  const int l    = lane & 15;
  const int mtile = blockIdx.x;
  const long nstart = (long)blockIdx.y * tilesPerChunk;

  const _Float16* arow = A16 + ((size_t)mtile * 16 + l) * kDim;
  v16h a0, a1;
  load_a64(arow, half, a0, a1);

  float accv[8] = {};
  for (int t = 0; t < tilesPerChunk; ++t) {
    // B 32x16 operand: lane holds column n=l, K = half*16 .. +16 (contiguous E-row bytes)
    const _Float16* brow = B16 + ((size_t)(nstart + t) * 16 + l) * kDim;
    v16h b0 = *(const v16h*)(brow + half * 16);
    v16h b1 = *(const v16h*)(brow + 32 + half * 16);
    v8f c = {};
    c = wmma_k32(a0, b0, c);
    c = wmma_k32(a1, b1, c);
#pragma unroll
    for (int r = 0; r < 8; ++r) accv[r] += __expf(c[r] * (1.0f / kTemp));
  }
  // row m = mtile*16 + half*8 + r lives across the 16 lanes of this half-wave
#pragma unroll
  for (int r = 0; r < 8; ++r) {
    float v = accv[r];
#pragma unroll
    for (int m = 8; m >= 1; m >>= 1) v += __shfl_xor(v, m, 16);
    if (l == 0) atomicAdd(&S[(size_t)mtile * 16 + half * 8 + r], v);
  }
}

// ---------------------------------------------------------------- y = ue @ pe^T  (1024x1024, K=64)
__global__ void __launch_bounds__(32)
k_ygemm(const _Float16* __restrict__ A16, const _Float16* __restrict__ B16,
        float* __restrict__ Y) {
  const int lane = threadIdx.x;
  const int half = lane >> 4;
  const int l    = lane & 15;
  const int mtile = blockIdx.x;
  const int ntile = blockIdx.y;

  const _Float16* arow = A16 + ((size_t)mtile * 16 + l) * kDim;
  v16h a0, a1;
  load_a64(arow, half, a0, a1);

  const _Float16* brow = B16 + ((size_t)ntile * 16 + l) * kDim;
  v16h b0 = *(const v16h*)(brow + half * 16);
  v16h b1 = *(const v16h*)(brow + 32 + half * 16);
  v8f c = {};
  c = wmma_k32(a0, b0, c);
  c = wmma_k32(a1, b1, c);
#pragma unroll
  for (int r = 0; r < 8; ++r)
    Y[((size_t)mtile * 16 + half * 8 + r) * kBatch + (size_t)ntile * 16 + l] = c[r];
}

// ---------------------------------------------------------------- mean of log(S + 1e-8)
__global__ void k_logmean(const float* __restrict__ S, int n, float* accSlot, float invn) {
  int i = blockIdx.x * blockDim.x + threadIdx.x;
  if (i < n) atomicAdd(accSlot, logf(S[i] + 1e-8f) * invn);
}

// ---------------------------------------------------------------- pos_score term: clip(dot(G,E[id])/TEMP, +-5), mean
__global__ void __launch_bounds__(32)
k_pos(const float* __restrict__ Gs, const float* __restrict__ E,
      const int* __restrict__ idxA, const int* __restrict__ idxB,
      float* accSlot, float invn) {
  int b = blockIdx.x, t = threadIdx.x;
  int id = (idxB != nullptr && b >= kBatch) ? idxB[b - kBatch] : idxA[b];
  float v = Gs[(size_t)b * kDim + t]      * E[(size_t)id * kDim + t] +
            Gs[(size_t)b * kDim + t + 32] * E[(size_t)id * kDim + t + 32];
#pragma unroll
  for (int m = 16; m >= 1; m >>= 1) v += __shfl_xor(v, m, 32);
  if (t == 0) {
    float s = v * (1.0f / kTemp);
    s = fminf(5.0f, fmaxf(-5.0f, s));
    atomicAdd(accSlot, s * invn);
  }
}

// ---------------------------------------------------------------- normalized sampled rows -> f16
__global__ void __launch_bounds__(32)
k_norm(const float* __restrict__ E, const int* __restrict__ idx,
       _Float16* __restrict__ out16) {
  int b = blockIdx.x, t = threadIdx.x;
  int id = idx[b];
  float x0 = E[(size_t)id * kDim + t];
  float x1 = E[(size_t)id * kDim + t + 32];
  float ss = x0 * x0 + x1 * x1;
#pragma unroll
  for (int m = 16; m >= 1; m >>= 1) ss += __shfl_xor(ss, m, 32);
  float inv = rsqrtf(ss);
  out16[(size_t)b * kDim + t]      = (_Float16)(x0 * inv);
  out16[(size_t)b * kDim + t + 32] = (_Float16)(x1 * inv);
}

// ---------------------------------------------------------------- loss_r per-row stabilized logsumexp with diag/col0 swap
__global__ void k_lossr(const float* __restrict__ Y, float* accSlot) {
  __shared__ float red[256];
  int b = blockIdx.x, tid = threadIdx.x;
  const float* row = Y + (size_t)b * kBatch;
  float dg = row[b];  // original diagonal -> y'[b,0]
  float c0 = row[0];  // original col 0    -> y'[b,b]
  float mx = -3.0e38f;
  for (int j = 1 + tid; j < kBatch; j += 256) {
    float v = (j == b) ? c0 : row[j];
    mx = fmaxf(mx, v);
  }
  red[tid] = mx; __syncthreads();
  for (int s = 128; s > 0; s >>= 1) { if (tid < s) red[tid] = fmaxf(red[tid], red[tid + s]); __syncthreads(); }
  mx = red[0]; __syncthreads();
  float sm = 0.0f;
  for (int j = 1 + tid; j < kBatch; j += 256) {
    float v = (j == b) ? c0 : row[j];
    sm += __expf((v - mx) * (1.0f / kT1));
  }
  red[tid] = sm; __syncthreads();
  for (int s = 128; s > 0; s >>= 1) { if (tid < s) red[tid] += red[tid + s]; __syncthreads(); }
  if (tid == 0) {
    float lse = mx * (1.0f / kT1) + logf(red[0]);
    float lr  = kT2 * lse - dg * (1.0f / kT1);
    atomicAdd(accSlot, lr * (1.0f / kBatch));
  }
}

// ---------------------------------------------------------------- sum of squares (regularizer)
__global__ void k_sumsq(const float* __restrict__ x, size_t n, float* accSlot) {
  size_t i  = (size_t)blockIdx.x * blockDim.x + threadIdx.x;
  size_t st = (size_t)gridDim.x * blockDim.x;
  float s = 0.0f;
  for (; i < n; i += st) { float v = x[i]; s += v * v; }
#pragma unroll
  for (int m = 16; m >= 1; m >>= 1) s += __shfl_xor(s, m, 32);
  if ((threadIdx.x & 31) == 0) atomicAdd(accSlot, s);
}

// ---------------------------------------------------------------- final combine -> (loss, loss_r, L1*loss_s)
__global__ void k_final(const float* __restrict__ acc, float* __restrict__ out) {
  float neg    = acc[0] + acc[1];
  float pos    = acc[2] + acc[3];
  float loss_s = neg - pos;
  float lr     = acc[4];
  float reg    = kL2 * acc[5];
  out[0] = lr + kL1 * loss_s + reg;
  out[1] = lr;
  out[2] = kL1 * loss_s;
}

// =================================================================
extern "C" void kernel_launch(void* const* d_in, const int* in_sizes, int n_in,
                              void* d_out, int out_size, void* d_ws, size_t ws_size,
                              hipStream_t stream) {
  (void)in_sizes; (void)n_in; (void)out_size; (void)ws_size;
  const float* Eu0  = (const float*)d_in[0];
  const float* Ei0  = (const float*)d_in[1];
  const float* ums  = (const float*)d_in[2];   // u_mul_s (N_USERS x 5)
  const float* vms  = (const float*)d_in[3];   // v_mul_s (N_ITEMS x 5)
  const float* ut   = (const float*)d_in[4];   // (5 x N_USERS)
  const float* vt   = (const float*)d_in[5];   // (5 x N_ITEMS)
  const float* vals = (const float*)d_in[6];
  const int*   rows = (const int*)d_in[7];
  const int*   cols = (const int*)d_in[8];
  const int*   uids = (const int*)d_in[9];
  const int*   pos  = (const int*)d_in[10];
  const int*   neg  = (const int*)d_in[11];
  float* out = (float*)d_out;

  // ---- workspace layout (floats; f16 region after; ~102 MB total)
  float* w = (float*)d_ws;
  size_t o = 0;
  float* zAu  = w + o; o += (size_t)kUsers * kDim;   // layer-1 z_u
  float* zAi  = w + o; o += (size_t)kItems * kDim;   // layer-1 z_i
  float* Eu   = w + o; o += (size_t)kUsers * kDim;   // layer-2 z_u -> final E_u
  float* Ei   = w + o; o += (size_t)kItems * kDim;   // layer-2 z_i -> final E_i
  float* tmpV = w + o; o += kRank * kDim;            // vt @ (E_i0 + zA_i)
  float* tmpU = w + o; o += kRank * kDim;            // ut @ (E_u0 + zA_u)
  float* Su   = w + o; o += kBatch;
  float* Si   = w + o; o += 2 * kBatch;
  float* acc  = w + o; o += 16;
  size_t zeroN = o;                                  // everything above needs zeroing
  float* Gus  = w + o; o += (size_t)kBatch * kDim;
  float* Gis  = w + o; o += (size_t)2 * kBatch * kDim;
  float* Y    = w + o; o += (size_t)kBatch * kBatch;
  _Float16* h = (_Float16*)(w + o);
  size_t ho = 0;
  _Float16* Eu16 = h + ho; ho += (size_t)kUsers * kDim;
  _Float16* Ei16 = h + ho; ho += (size_t)kItems * kDim;
  _Float16* Gu16 = h + ho; ho += (size_t)kBatch * kDim;
  _Float16* Gi16 = h + ho; ho += (size_t)2 * kBatch * kDim;
  _Float16* ue16 = h + ho; ho += (size_t)kBatch * kDim;
  _Float16* pe16 = h + ho; ho += (size_t)kBatch * kDim;

  // 1) zero accumulators
  k_zero<<<4096, 256, 0, stream>>>(w, zeroN);

  // 2) layer-1 SpMM (e = E0), 3) layer-2 SpMM (e = zA)
  const long spmmThreads = (long)kNnz * kDim;
  const int  spmmBlocks  = (int)((spmmThreads + 255) / 256);
  k_spmm<<<spmmBlocks, 256, 0, stream>>>(Eu0, Ei0, rows, cols, vals, zAu, zAi);
  k_spmm<<<spmmBlocks, 256, 0, stream>>>(zAu, zAi, rows, cols, vals, Eu, Ei);

  // 4) rank-5 temps (sum of both layers' low-rank inputs)
  k_rank<<<512, 256, 0, stream>>>(vt, Ei0, zAi, tmpV, kItems);
  k_rank<<<512, 256, 0, stream>>>(ut, Eu0, zAu, tmpU, kUsers);

  // 5) final E (in place) + f16 mirrors for WMMA B operands
  k_combine<<<4096, 256, 0, stream>>>(Eu0, zAu, Eu, Eu16, (size_t)kUsers * kDim);
  k_combine<<<4096, 256, 0, stream>>>(Ei0, zAi, Ei, Ei16, (size_t)kItems * kDim);

  // 6) gather sampled G rows (G never materialized in full)
  k_gather<<<kBatch, 64, 0, stream>>>(Eu0, ums, tmpV, uids, nullptr, Gus, Gu16);
  k_gather<<<2 * kBatch, 64, 0, stream>>>(Ei0, vms, tmpU, pos, neg, Gis, Gi16);

  // 7) fused GEMM + exp-sum:  user: 64 m-tiles x 6250 n-tiles (125 chunks of 50)
  //                           item: 128 m-tiles x 3125 n-tiles (125 chunks of 25)
  k_expsum<<<dim3(64, 125), 32, 0, stream>>>(Gu16, Eu16, Su, 50);
  k_expsum<<<dim3(128, 125), 32, 0, stream>>>(Gi16, Ei16, Si, 25);

  // 8) neg_score = mean log(S + 1e-8)  (user + item halves into acc[0], acc[1])
  k_logmean<<<4, 256, 0, stream>>>(Su, kBatch, acc + 0, 1.0f / kBatch);
  k_logmean<<<8, 256, 0, stream>>>(Si, 2 * kBatch, acc + 1, 1.0f / (2 * kBatch));

  // 9) pos_score terms into acc[2], acc[3]
  k_pos<<<kBatch, 32, 0, stream>>>(Gus, Eu, uids, nullptr, acc + 2, 1.0f / kBatch);
  k_pos<<<2 * kBatch, 32, 0, stream>>>(Gis, Ei, pos, neg, acc + 3, 1.0f / (2 * kBatch));

  // 10) normalized sampled embeddings, 11) y = ue @ pe^T via WMMA
  k_norm<<<kBatch, 32, 0, stream>>>(Eu, uids, ue16);
  k_norm<<<kBatch, 32, 0, stream>>>(Ei, pos, pe16);
  k_ygemm<<<dim3(64, 64), 32, 0, stream>>>(ue16, pe16, Y);

  // 12) loss_r into acc[4]
  k_lossr<<<kBatch, 256, 0, stream>>>(Y, acc + 4);

  // 13) regularizer into acc[5]
  k_sumsq<<<2048, 256, 0, stream>>>(Eu0, (size_t)kUsers * kDim, acc + 5);
  k_sumsq<<<1024, 256, 0, stream>>>(Ei0, (size_t)kItems * kDim, acc + 5);

  // 14) combine -> (loss, loss_r, L1*loss_s)
  k_final<<<1, 1, 0, stream>>>(acc, out);
}
